// DSRGLayer_11081015623888
// MI455X (gfx1250) — matrix operand order repaired and checked by reading, not compiled, not run on hardware
//
#include <hip/hip_runtime.h>
#include <math.h>

// ---------------- problem constants (fixed by reference) ----------------
static constexpr int Bn = 4, Cn = 21, Hn = 321, Wn = 321;
static constexpr int HW = Hn * Wn;
static constexpr int NPIX = Bn * HW;            // 412164
static constexpr int SEG  = (Wn + 15) / 16;     // 21 x-segments of 16 px
static constexpr int YG   = (Hn + 3) / 4;       // 81 groups of 4 rows

typedef __attribute__((ext_vector_type(16))) __bf16 v16bf;
typedef __attribute__((ext_vector_type(8)))  float  v8f;

// Q tile: (21+1 zero pad class) x (12+1 guard) rows x 24 cols, flat stride 312
// I tile: 3 x (12+1 guard) rows x 24 cols (guard keeps cw finite for K-pad)
static constexpr int QROWS = 13;
static constexpr int CSTR  = QROWS * 24;             // 312 floats per class/ch
static constexpr int QTILE = (Cn + 1) * CSTR;        // 6864 floats
static constexpr int ITILE = 3 * CSTR;               // 936  floats
static constexpr int POOL  = QTILE + ITILE;          // 7800 (mult of 4)

// async global->LDS copy of one dword (gfx1250, tracked by ASYNCcnt)
__device__ __forceinline__ void async_ld_b32(unsigned lds_off, const float* g) {
  asm volatile("global_load_async_to_lds_b32 %0, %1, off"
               :: "v"(lds_off), "v"(g) : "memory");
}
__device__ __forceinline__ void wait_async0() {
  asm volatile("s_wait_asynccnt 0x0" ::: "memory");
}

// ======================= CRF: init (logsm + Q0) =========================
__global__ void k_init(const float* __restrict__ sm,
                       float* __restrict__ Q0, float* __restrict__ LOGS) {
  int p = blockIdx.x * blockDim.x + threadIdx.x;
  if (p >= NPIX) return;
  int b = p / HW, rem = p - b * HW;
  float l[Cn]; float mx = -1e30f;
#pragma unroll
  for (int c = 0; c < Cn; ++c) {
    size_t qi = (size_t)(b * Cn + c) * HW + rem;
    float v = __logf(fmaxf(sm[qi], 1e-5f));
    LOGS[qi] = v; l[c] = v; mx = fmaxf(mx, v);
  }
  float s = 0.f;
#pragma unroll
  for (int c = 0; c < Cn; ++c) { l[c] = __expf(l[c] - mx); s += l[c]; }
  float inv = 1.f / s;
#pragma unroll
  for (int c = 0; c < Cn; ++c)
    Q0[(size_t)(b * Cn + c) * HW + rem] = l[c] * inv;
}

// ============== CRF: one mean-field iteration (WMMA core) ===============
// Block = 4 waves = 4 consecutive rows of one 16-px segment.
// Window 9x24, K linearized & padded to 224 = 7x32; LDS offset affine in K:
//   off(c, K) = c*312 + wv*24 + K  -> all fragment loads are contiguous.
// Geometry-dependent weights precomputed per block into LDS tables.
__global__ __launch_bounds__(128) void k_crf_iter(
    const float* __restrict__ Qsrc, float* __restrict__ Qdst,
    const float* __restrict__ LOGS, const float* __restrict__ img) {
  __shared__ __align__(16) float pool[POOL];            // Qs | Is, zero-filled
  __shared__ float msg[4][16][Cn][2];                   // [wave][px][class][2]
  __shared__ __align__(32) unsigned bgtab[7][32][8];    // Gaussian B, bf16x2
  __shared__ __align__(16) float wbmtab[7][32][16];     // bilateral mask f32

  int tid = threadIdx.x;
  // ---- zero pool ----
  float4* p4 = (float4*)pool;
  for (int i = tid; i < POOL / 4; i += 128) p4[i] = make_float4(0.f, 0.f, 0.f, 0.f);
  // ---- build Gaussian B fragments (static per lane/e/kc) ----
  for (int d = tid; d < 7 * 256; d += 128) {
    int kc = d >> 8, r = d & 255, ln = r >> 3, j = r & 7;
    unsigned pk = 0;
#pragma unroll
    for (int t = 0; t < 2; ++t) {
      int e = 2 * j + t;
      int hi = ln >> 4, N = ln & 15;
      int w = kc * 32 + e + hi * 16;
      int wy = w / 24, wx = w - wy * 24;
      int dy = wy - 4, dxv = wx - 4 - N;
      bool ok = (dxv >= -4) && (dxv <= 4) && !(dy == 0 && dxv == 0);
      float g = ok ? __expf(-(float)(dy * dy + dxv * dxv) * (1.f / 18.f)) : 0.f;
      unsigned short hb = __builtin_bit_cast(unsigned short, (__bf16)g);
      pk |= ((unsigned)hb) << (16 * t);
    }
    ((unsigned*)bgtab)[d] = pk;
  }
  // ---- build bilateral weight masks (static per lane/e/kc) ----
  for (int f = tid; f < 7 * 512; f += 128) {
    int kc = f >> 9, r = f & 511, ln = r >> 4, e = r & 15;
    int hi = ln >> 4, N = ln & 15;
    int w = kc * 32 + e + hi * 16;
    int wy = w / 24, wx = w - wy * 24;
    int dy = wy - 4, dxv = wx - 4 - N;
    bool ok = (dxv >= -4) && (dxv <= 4) && !(dy == 0 && dxv == 0);
    ((float*)wbmtab)[f] =
        ok ? __expf(-(float)(dy * dy + dxv * dxv) * (1.f / 4802.f)) : 0.f;
  }

  // tile decode
  int blk = blockIdx.x;
  int b   = blk / (YG * SEG);
  int rm  = blk - b * (YG * SEG);
  int y0  = (rm / SEG) * 4;
  int x0  = (rm - (rm / SEG) * SEG) * 16;
  __syncthreads();

  // ---- async stage: Q window (21 x 12 x 24) ----
  for (int s = tid; s < Cn * 288; s += 128) {
    int c = s / 288, r2 = s - c * 288;
    int r = r2 / 24, wx = r2 - r * 24;
    int qy = y0 - 4 + r, qx = x0 - 4 + wx;
    if (qy >= 0 && qy < Hn && qx >= 0 && qx < Wn) {
      unsigned loff = (unsigned)(uintptr_t)&pool[c * CSTR + r * 24 + wx];
      async_ld_b32(loff, Qsrc + ((size_t)(b * Cn + c) * Hn + qy) * Wn + qx);
    }
  }
  // ---- async stage: image window (3 x 12 x 24) ----
  for (int s = tid; s < 3 * 288; s += 128) {
    int ch = s / 288, r2 = s - ch * 288;
    int r = r2 / 24, wx = r2 - r * 24;
    int qy = y0 - 4 + r, qx = x0 - 4 + wx;
    if (qy >= 0 && qy < Hn && qx >= 0 && qx < Wn) {
      unsigned loff = (unsigned)(uintptr_t)&pool[QTILE + ch * CSTR + r * 24 + wx];
      async_ld_b32(loff, img + ((size_t)(b * 3 + ch) * Hn + qy) * Wn + qx);
    }
  }
  wait_async0();
  __syncthreads();

  int lane = tid & 31, wv = tid >> 5;
  int N = lane & 15, hi = (lane >> 4) & 1;
  int y = y0 + wv;
  bool active = y < Hn;

  if (active) {
    const float* QsF = pool;
    const float* IsF = pool + QTILE;
    int c0 = N;
    int c1 = N + 16; c1 = (c1 < Cn) ? c1 : Cn;          // zero pad class
    float Ic0 = IsF[0 * CSTR + (wv + 4) * 24 + N + 4];
    float Ic1 = IsF[1 * CSTR + (wv + 4) * 24 + N + 4];
    float Ic2 = IsF[2 * CSTR + (wv + 4) * 24 + N + 4];

    // kc-independent base offsets (kc*32 folds into load immediates)
    const float* pa0 = QsF + c0 * CSTR + wv * 24 + hi * 16;
    const float* pa1 = QsF + c1 * CSTR + wv * 24 + hi * 16;
    const float* pi0 = IsF + 0 * CSTR + wv * 24 + hi * 16;
    const float* pi1 = IsF + 1 * CSTR + wv * 24 + hi * 16;
    const float* pi2 = IsF + 2 * CSTR + wv * 24 + hi * 16;

    v8f dg0 = {0,0,0,0,0,0,0,0}, dg1 = {0,0,0,0,0,0,0,0};
    v8f db0 = {0,0,0,0,0,0,0,0}, db1 = {0,0,0,0,0,0,0,0};

#pragma unroll
    for (int kc = 0; kc < 7; ++kc) {
      int k32 = kc * 32;
      // ---- A fragments: 4 contiguous float4 loads per class tile ----
      v16bf a0, a1;
      {
        float q[16], qq[16];
        *(float4*)&q[0]   = *(const float4*)(pa0 + k32);
        *(float4*)&q[4]   = *(const float4*)(pa0 + k32 + 4);
        *(float4*)&q[8]   = *(const float4*)(pa0 + k32 + 8);
        *(float4*)&q[12]  = *(const float4*)(pa0 + k32 + 12);
        *(float4*)&qq[0]  = *(const float4*)(pa1 + k32);
        *(float4*)&qq[4]  = *(const float4*)(pa1 + k32 + 4);
        *(float4*)&qq[8]  = *(const float4*)(pa1 + k32 + 8);
        *(float4*)&qq[12] = *(const float4*)(pa1 + k32 + 12);
#pragma unroll
        for (int e = 0; e < 16; ++e) { a0[e] = (__bf16)q[e]; a1[e] = (__bf16)qq[e]; }
      }
      // ---- Gaussian B: direct packed-bf16 fragment load ----
      v16bf bg = *(const v16bf*)&bgtab[kc][lane][0];
      // ---- bilateral B: mask table * exp(color distance) ----
      v16bf bb;
      {
        float wm[16], cA[16], cB[16], cC[16];
        const float4* wmp = (const float4*)&wbmtab[kc][lane][0];
#pragma unroll
        for (int j = 0; j < 4; ++j) {
          *(float4*)&wm[4 * j] = wmp[j];
          *(float4*)&cA[4 * j] = *(const float4*)(pi0 + k32 + 4 * j);
          *(float4*)&cB[4 * j] = *(const float4*)(pi1 + k32 + 4 * j);
          *(float4*)&cC[4 * j] = *(const float4*)(pi2 + k32 + 4 * j);
        }
#pragma unroll
        for (int e = 0; e < 16; ++e) {
          float d0 = Ic0 - cA[e], d1 = Ic1 - cB[e], d2v = Ic2 - cC[e];
          float cw = __expf(-(d0 * d0 + d1 * d1 + d2v * d2v) * (1.f / 50.f));
          bb[e] = (__bf16)(wm[e] * cw);
        }
      }
      dg0 = __builtin_amdgcn_wmma_f32_16x16x32_bf16(false, a0, false, bg, (short)0, dg0, false, false);
      dg1 = __builtin_amdgcn_wmma_f32_16x16x32_bf16(false, a1, false, bg, (short)0, dg1, false, false);
      db0 = __builtin_amdgcn_wmma_f32_16x16x32_bf16(false, a0, false, bb, (short)0, db0, false, false);
      db1 = __builtin_amdgcn_wmma_f32_16x16x32_bf16(false, a1, false, bb, (short)0, db1, false, false);
    }
    // ---- scatter D tiles (M = r + 8*hi, N = lane&15) to LDS ----
#pragma unroll
    for (int r = 0; r < 8; ++r) {
      int cc0 = r + hi * 8;
      msg[wv][N][cc0][0] = dg0[r];
      msg[wv][N][cc0][1] = db0[r];
      int cc1 = 16 + cc0;
      if (cc1 < Cn) { msg[wv][N][cc1][0] = dg1[r]; msg[wv][N][cc1][1] = db1[r]; }
    }
  }
  __syncthreads();
  // ---- per-pixel softmax: Q' = softmax(logsm + 3*mg + 4*mb) ----
  if (active && lane < 16) {
    int x = x0 + lane;
    if (x < Wn) {
      float vals[Cn]; float mx = -1e30f;
#pragma unroll
      for (int c = 0; c < Cn; ++c) {
        size_t qi = ((size_t)(b * Cn + c) * Hn + y) * Wn + x;
        float v = LOGS[qi] + 3.f * msg[wv][lane][c][0] + 4.f * msg[wv][lane][c][1];
        vals[c] = v; mx = fmaxf(mx, v);
      }
      float s = 0.f;
#pragma unroll
      for (int c = 0; c < Cn; ++c) { vals[c] = __expf(vals[c] - mx); s += vals[c]; }
      float inv = 1.f / s;
#pragma unroll
      for (int c = 0; c < Cn; ++c)
        Qdst[((size_t)(b * Cn + c) * Hn + y) * Wn + x] = vals[c] * inv;
    }
  }
}

// ============ CRF: transpose (C,H,W)->(H,W,C), clamp, renorm ============
__global__ void k_crf_out(const float* __restrict__ Qf, float* __restrict__ out) {
  int p = blockIdx.x * blockDim.x + threadIdx.x;
  if (p >= NPIX) return;
  int b = p / HW, rem = p - b * HW;
  float v[Cn]; float s = 0.f;
#pragma unroll
  for (int c = 0; c < Cn; ++c) {
    float x = fmaxf(Qf[(size_t)(b * Cn + c) * HW + rem], 1e-5f);
    v[c] = x; s += x;
  }
  float inv = 1.f / s;
#pragma unroll
  for (int c = 0; c < Cn; ++c) out[(size_t)p * Cn + c] = v[c] * inv;
}

// ============== seed: per-pixel init (seedbits, label_map, good) ========
__global__ void k_seed_init(const float* __restrict__ cues,
                            const int* __restrict__ labels,
                            const float* __restrict__ sm,
                            unsigned* __restrict__ LBL,
                            unsigned* __restrict__ SEEDB,
                            unsigned* __restrict__ REACH) {
  int p = blockIdx.x * blockDim.x + threadIdx.x;
  if (p >= NPIX) return;
  int b = p / HW, rem = p - b * HW;
  float sc[Cn]; float mx = 0.f;
#pragma unroll
  for (int c = 0; c < Cn; ++c) {
    float s = cues[(size_t)(b * Cn + c) * HW + rem] * (float)labels[b * Cn + c];
    s = (s > 0.5f) ? s : 0.f;
    sc[c] = s; mx = fmaxf(mx, s);
  }
  unsigned bits = 0;
#pragma unroll
  for (int c = 0; c < Cn; ++c)
    if (sc[c] == mx && sc[c] != 0.f) bits |= (1u << c);
  float bp = -3.4e38f; int bc = 0;
#pragma unroll
  for (int c = 0; c < Cn; ++c) {
    float m = (labels[b * Cn + c] == 1)
                  ? fmaxf(sm[(size_t)(b * Cn + c) * HW + rem], 1e-5f)
                  : -3.4e38f;
    if (m > bp) { bp = m; bc = c; }
  }
  unsigned lm0 = bits ? (unsigned)(31 - __clz(bits)) : 255u;  // cmax else 255
  unsigned lm = (bp > 0.85f) ? (unsigned)bc : lm0;
  bool good = (lm < (unsigned)Cn) && ((bits >> lm) & 1u);
  LBL[p] = lm; SEEDB[p] = bits; REACH[p] = good ? 1u : 0u;
}

// ====== seed: bitmask flood fill (reach within equal-label regions) =====
__global__ __launch_bounds__(32) void k_reach(const unsigned* __restrict__ LBL,
                                              unsigned* __restrict__ REACH) {
  int b = blockIdx.z;
  int tx0 = blockIdx.x * 30 - 1;
  int gy  = blockIdx.y * 30 - 1 + (int)threadIdx.x;
  int lane = threadIdx.x;
  bool rowok = (gy >= 0 && gy < Hn);
  int base = b * HW + gy * Wn;
  unsigned lbl[32]; unsigned valid = 0, r = 0;
#pragma unroll
  for (int x = 0; x < 32; ++x) {
    int gx = tx0 + x;
    unsigned l = 0xFFFFFFFFu, rb = 0;
    if (rowok && gx >= 0 && gx < Wn) { l = LBL[base + gx]; rb = REACH[base + gx]; }
    lbl[x] = l;
    if (l < (unsigned)Cn) valid |= (1u << x);
    if (rb) r |= (1u << x);
  }
  unsigned eqL = 0, eqR = 0, eqU = 0, eqD = 0;
#pragma unroll
  for (int x = 0; x < 32; ++x) {
    bool v = lbl[x] < (unsigned)Cn;
    if (x > 0  && v && lbl[x] == lbl[x - 1]) eqL |= (1u << x);
    if (x < 31 && v && lbl[x] == lbl[x + 1]) eqR |= (1u << x);
    unsigned up = (unsigned)__shfl_up((int)lbl[x], 1);
    unsigned dn = (unsigned)__shfl_down((int)lbl[x], 1);
    if (lane > 0  && v && up == lbl[x]) eqU |= (1u << x);
    if (lane < 31 && v && dn == lbl[x]) eqD |= (1u << x);
  }
  r &= valid;
  for (int it = 0; it < 64; ++it) {
    unsigned m = r;
    m |= (r << 1) & eqL;
    m |= (r >> 1) & eqR;
    unsigned u = (unsigned)__shfl_up((int)r, 1);   if (lane == 0)  u = 0;
    unsigned d = (unsigned)__shfl_down((int)r, 1); if (lane == 31) d = 0;
    m |= u & eqU;
    m |= d & eqD;
    r = m & valid;
  }
  if (rowok && lane >= 1 && lane <= 30) {
#pragma unroll
    for (int x = 1; x <= 30; ++x) {
      int gx = tx0 + x;
      if (gx >= 0 && gx < Wn && ((r >> x) & 1u)) REACH[base + gx] = 1u;
    }
  }
}

// =================== seed: finalize + argmax -> int out =================
__global__ void k_seed_final(const int* __restrict__ labels,
                             const unsigned* __restrict__ LBL,
                             const unsigned* __restrict__ SEEDB,
                             const unsigned* __restrict__ REACH,
                             int* __restrict__ out) {
  int p = blockIdx.x * blockDim.x + threadIdx.x;
  if (p >= NPIX) return;
  int b = p / HW;
  unsigned lm = LBL[p], sb = SEEDB[p];
  if (lm < (unsigned)Cn) {
    unsigned scls = (sb >> lm) & 1u;
    int ssum = __popc(sb);
    bool excl = (ssum == 1) && !scls;           // lab_eff = SENT pixels
    bool keep = (REACH[p] != 0) && !excl;
    if (labels[b * Cn + (int)lm] == 1 && keep) sb |= (1u << lm);
  }
  out[p] = sb ? (__ffs((int)sb) - 1) : 255;     // first max index, else 255
}

// =============================== launch =================================
extern "C" void kernel_launch(void* const* d_in, const int* in_sizes, int n_in,
                              void* d_out, int out_size, void* d_ws, size_t ws_size,
                              hipStream_t stream) {
  const float* im      = (const float*)d_in[0];  // (B,3,H,W)
  const int*   labels  = (const int*)d_in[1];    // (B,C)
  const float* cues    = (const float*)d_in[2];  // (B,C,H,W)
  const float* softmax = (const float*)d_in[3];  // (B,C,H,W)

  int* outSeed = (int*)d_out;                    // (B,H,W) int32
  float* outCrf = (float*)d_out + NPIX;          // (B,H,W,C) f32

  // workspace (~109 MB): Q ping/pong + logsm + 3 int planes
  size_t QN = (size_t)Bn * Cn * HW;
  float* Q0   = (float*)d_ws;
  float* Q1   = Q0 + QN;
  float* LOGS = Q1 + QN;
  unsigned* LBL   = (unsigned*)(LOGS + QN);
  unsigned* SEEDB = LBL + NPIX;
  unsigned* REACH = SEEDB + NPIX;

  dim3 pixGrid((NPIX + 255) / 256), pixBlk(256);

  // ---- CRF ----
  k_init<<<pixGrid, pixBlk, 0, stream>>>(softmax, Q0, LOGS);
  dim3 crfGrid(Bn * YG * SEG), crfBlk(128);
  float* src = Q0; float* dst = Q1;
  for (int it = 0; it < 5; ++it) {
    k_crf_iter<<<crfGrid, crfBlk, 0, stream>>>(src, dst, LOGS, im);
    float* tmp = src; src = dst; dst = tmp;
  }
  k_crf_out<<<pixGrid, pixBlk, 0, stream>>>(src, outCrf);  // src = final Q

  // ---- seed / region growing ----
  k_seed_init<<<pixGrid, pixBlk, 0, stream>>>(cues, labels, softmax, LBL, SEEDB, REACH);
  dim3 rGrid((Wn + 29) / 30, (Hn + 29) / 30, Bn);
  for (int pass = 0; pass < 16; ++pass)
    k_reach<<<rGrid, 32, 0, stream>>>(LBL, REACH);
  k_seed_final<<<pixGrid, pixBlk, 0, stream>>>(labels, LBL, SEEDB, REACH, outSeed);
}
